// Refine_54305566491106
// MI455X (gfx1250) — compile-verified, matrix operand order checked by reading, not run on hardware
//
#include <hip/hip_runtime.h>
#include <hip/hip_bf16.h>

typedef __attribute__((ext_vector_type(16))) _Float16 v16h;
typedef __attribute__((ext_vector_type(8)))  _Float16 v8h;
typedef __attribute__((ext_vector_type(8)))  float    v8f;
typedef int v4i_vs __attribute__((vector_size(16)));   // matches builtin's V4i

// ---------------------------------------------------------------------------
// CDNA5 async Global->LDS staging (ASYNCcnt-tracked DMA, no VGPR round-trip).
// Builtin signature (from hipcc diagnostic): (V4i AS1*, V4i AS3*, Imm, Imm).
// Falls back to a manual copy if the builtin is absent on this toolchain.
// ---------------------------------------------------------------------------
#if defined(__AMDGCN__) && __has_builtin(__builtin_amdgcn_global_load_async_to_lds_b128)
#define ASYNC_LDS 1
#else
#define ASYNC_LDS 0
#endif

__device__ __forceinline__ void stage16(_Float16* dst, const _Float16* src) {
#if ASYNC_LDS
  __builtin_amdgcn_global_load_async_to_lds_b128(
      (__attribute__((address_space(1))) v4i_vs*)(void*)src,
      (__attribute__((address_space(3))) v4i_vs*)(void*)dst, 0, 0);
#else
  *(v8h*)dst = *(const v8h*)src;
#endif
}

__device__ __forceinline__ void stage_wait_sync() {
#if ASYNC_LDS
#if __has_builtin(__builtin_amdgcn_s_wait_asynccnt)
  __builtin_amdgcn_s_wait_asynccnt(0);
#else
  asm volatile("s_wait_asynccnt 0x0" ::: "memory");
#endif
#endif
  __syncthreads();
}

// ---------------------------------------------------------------------------
// WMMA helpers (CDNA5 wave32, V_WMMA_F32_16X16X32_F16)
// ---------------------------------------------------------------------------
__device__ __forceinline__ v8f wmma32(v16h a, v16h b, v8f c) {
  return __builtin_amdgcn_wmma_f32_16x16x32_f16(
      /*neg_a=*/false, a, /*neg_b=*/false, b,
      /*c_mod=*/(short)0, c, /*reuse_a=*/false, /*reuse_b=*/false);
}

// Load a 16(row) x 32(k) f16 fragment from LDS, rows `stride` halfs apart.
// ISA 7.12.2 layout: lanes 0-15 row=lane, K 0-7 & 16-23; lanes 16-31
// row=lane-16, K 8-15 & 24-31. Two ds_load_b128 per lane.
__device__ __forceinline__ v16h ld_frag(const _Float16* base, int stride) {
  const int lane = threadIdx.x & 31;
  const int r  = lane & 15;
  const int kh = (lane >> 4) << 3;  // 0 or 8
  const _Float16* p = base + r * stride + kh;
  v8h lo = *(const v8h*)(p);
  v8h hi = *(const v8h*)(p + 16);
  v16h out;
#pragma unroll
  for (int i = 0; i < 8; ++i) { out[i] = lo[i]; out[i + 8] = hi[i]; }
  return out;
}

// ---------------------------------------------------------------------------
// Pack kernels
// ---------------------------------------------------------------------------
__global__ void f32_to_f16_kernel(const float* __restrict__ src,
                                  _Float16* __restrict__ dst, int n) {
  for (int i = blockIdx.x * blockDim.x + threadIdx.x; i < n;
       i += gridDim.x * blockDim.x)
    dst[i] = (_Float16)src[i];
}

// feature NCHW [8][64][128][128] f32 -> channels-last [8][128][128][64] f16
__global__ void pack_feat_kernel(const float* __restrict__ src,
                                 _Float16* __restrict__ dst) {
  int i = blockIdx.x * blockDim.x + threadIdx.x;
  if (i >= 8 * 128 * 128 * 64) return;
  int c = i & 63;
  int x = (i >> 6) & 127;
  int y = (i >> 13) & 127;
  int b = i >> 20;
  dst[i] = (_Float16)src[((b * 64 + c) * 128 + y) * 128 + x];
}

// w1 [256][64][3][3] f32  ->  w1t [n=256][k=tap*64+c] f16
__global__ void pack_w1_kernel(const float* __restrict__ w1,
                               _Float16* __restrict__ w1t) {
  int i = blockIdx.x * blockDim.x + threadIdx.x;
  if (i >= 256 * 576) return;
  int n = i / 576, rem = i - n * 576;
  int tap = rem >> 6, c = rem & 63;                 // rem = tap*64 + c
  w1t[i] = (_Float16)w1[n * 576 + c * 9 + tap];
}

// ---------------------------------------------------------------------------
// Fused conv1(3x3,64->256)+ReLU+conv2(1x1,256->64) implicit-GEMM kernel.
// One block = 64 consecutive pixels on one image row. Halo slab staged once;
// WMMA A-fragments read straight from it (stride 72). w1 K-slices are
// double-buffered and staged with async Global->LDS DMA (1 barrier / step).
// GEMM1: M=64 x N=256 x K=576 (18 steps, 8 WMMA/wave/step).
// GEMM2: M=64 x N=64 x K=256 from relu(y1) staged f16 in LDS.
// ---------------------------------------------------------------------------
#define SLAB_HALFS  (3 * 66 * 72)               // 14256
#define SB0_OFF     SLAB_HALFS                  // w1 slice buf0: 256 x 40
#define SB1_OFF     (SLAB_HALFS + 256 * 40)     // w1 slice buf1
#define SY_OFF      (SLAB_HALFS + 2 * 256 * 40) // 34736: relu(y1): 64 x 264
#define SMEM_HALFS  (SY_OFF + 64 * 264)         // 51632 halfs = 103264 B

__global__ __launch_bounds__(256)
void conv_fused_kernel(const _Float16* __restrict__ featCL,
                       const _Float16* __restrict__ w1t,
                       const _Float16* __restrict__ w2t,
                       const float* __restrict__ bias1,
                       const float* __restrict__ bias2,
                       float* __restrict__ fOut) {
  __shared__ _Float16 smem[SMEM_HALFS];
  _Float16* slab = smem;                 // dead after GEMM1
  _Float16* sB0  = smem + SB0_OFF;       // dead after GEMM1
  _Float16* sB1  = smem + SB1_OFF;       // dead after GEMM1
  _Float16* sW2  = smem;                 // overlays slab/sB for GEMM2
  _Float16* sY   = smem + SY_OFF;

  const int tid = threadIdx.x;
  const int bid = blockIdx.x;                 // 2048 blocks = 8*128*2
  const int x0 = (bid & 1) * 64;
  const int y  = (bid >> 1) & 127;
  const int b  = bid >> 8;
  const int wave = tid >> 5;

  // ---- Stage halo slab: rows y-1..y+1, x-slots x0-1..x0+64, 64 ch, zero pad.
#pragma unroll
  for (int i = 0; i < 7; ++i) {
    int id = tid + 256 * i;                   // chunk of 8 halfs; 1584 chunks
    if (id < 3 * 66 * 8) {
      int slot = id >> 3, co = id & 7;
      int row = slot / 66, xs = slot - row * 66;
      int yy = y + row - 1, xx = x0 + xs - 1;
      v8h v = {};
      if (yy >= 0 && yy < 128 && xx >= 0 && xx < 128)
        v = *(const v8h*)(featCL + (((size_t)((b * 128 + yy) * 128 + xx)) << 6) + co * 8);
      *(v8h*)(slab + slot * 72 + co * 8) = v;
    }
  }

  const int bc8 = tid & 3;                    // w1 slice staging coords
  const int brow0 = tid >> 2;
  auto stageB = [&](int s, _Float16* buf) {
#pragma unroll
    for (int i = 0; i < 4; ++i) {
      int row = brow0 + 64 * i;               // 256 rows x 4 chunks
      stage16(buf + row * 40 + bc8 * 8, w1t + row * 576 + s * 32 + bc8 * 8);
    }
  };
  stageB(0, sB0);

  // wave owns N columns [wave*32, wave*32+32): acc[mt*2 + ntSub], mt 0..3
  v8f acc[8] = {v8f{}, v8f{}, v8f{}, v8f{}, v8f{}, v8f{}, v8f{}, v8f{}};

  for (int s = 0; s < 18; ++s) {
    stage_wait_sync();
    _Float16* cur = (s & 1) ? sB1 : sB0;
    if (s + 1 < 18) stageB(s + 1, (s & 1) ? sB0 : sB1);

    const int tap = s >> 1;
    const int dy = tap / 3, dx = tap % 3;     // slab-relative (0..2)
    const int cbase = (s & 1) * 32;
    const _Float16* aBase = slab + (dy * 66 + dx) * 72 + cbase;
    v16h a0 = ld_frag(aBase,            72);
    v16h a1 = ld_frag(aBase + 16 * 72,  72);
    v16h a2 = ld_frag(aBase + 32 * 72,  72);
    v16h a3 = ld_frag(aBase + 48 * 72,  72);
    v16h bf0 = ld_frag(cur + (wave * 32) * 40, 40);
    v16h bf1 = ld_frag(cur + (wave * 32 + 16) * 40, 40);
    acc[0] = wmma32(a0, bf0, acc[0]);
    acc[1] = wmma32(a0, bf1, acc[1]);
    acc[2] = wmma32(a1, bf0, acc[2]);
    acc[3] = wmma32(a1, bf1, acc[3]);
    acc[4] = wmma32(a2, bf0, acc[4]);
    acc[5] = wmma32(a2, bf1, acc[5]);
    acc[6] = wmma32(a3, bf0, acc[6]);
    acc[7] = wmma32(a3, bf1, acc[7]);
  }

  __syncthreads();   // slab/sB dead from here; safe to overlay with sW2

  // bias + ReLU -> sY (f16); meanwhile stage w2t (64x256) -> sW2 (stride 264)
  {
    const int lane = tid & 31;
    const int nLoc = lane & 15;
    const int mOff = (lane >> 4) * 8;
#pragma unroll
    for (int t = 0; t < 8; ++t) {
      int mt = t >> 1, nt = t & 1;
      int nIdx = wave * 32 + nt * 16 + nLoc;
      float bv = bias1[nIdx];
#pragma unroll
      for (int r = 0; r < 8; ++r) {
        float v = acc[t][r] + bv;
        v = v > 0.0f ? v : 0.0f;
        sY[(mt * 16 + mOff + r) * 264 + nIdx] = (_Float16)v;
      }
    }
  }
#pragma unroll
  for (int i = 0; i < 8; ++i) {
    int id = tid + 256 * i;                   // 2048 chunks of 8 halfs
    int row = id >> 5, c8 = id & 31;
    *(v8h*)(sW2 + row * 264 + c8 * 8) = *(const v8h*)(w2t + row * 256 + c8 * 8);
  }
  __syncthreads();

  // GEMM2: wave -> col tile nt2 = wave&3; m-tiles {wave>>2, (wave>>2)+2}
  const int nt2 = wave & 3;
  const int mtA = wave >> 2;
  v8f acc2[2] = {v8f{}, v8f{}};
#pragma unroll
  for (int ks = 0; ks < 8; ++ks) {
    v16h bb = ld_frag(sW2 + nt2 * 16 * 264 + ks * 32, 264);
    v16h a0 = ld_frag(sY + (mtA * 16) * 264 + ks * 32, 264);
    v16h a1 = ld_frag(sY + ((mtA + 2) * 16) * 264 + ks * 32, 264);
    acc2[0] = wmma32(a0, bb, acc2[0]);
    acc2[1] = wmma32(a1, bb, acc2[1]);
  }
  {
    const int lane = tid & 31;
    const int ch = nt2 * 16 + (lane & 15);
    const int mOff = (lane >> 4) * 8;
    float bv = bias2[ch];
#pragma unroll
    for (int t = 0; t < 2; ++t) {
      int mt = mtA + t * 2;
#pragma unroll
      for (int r = 0; r < 8; ++r) {
        int x = x0 + mt * 16 + mOff + r;
        fOut[((b * 64 + ch) * 128 + y) * 128 + x] = acc2[t][r] + bv;
      }
    }
  }
}

// ---------------------------------------------------------------------------
// Bilinear sampler: one thread per (p, j), loops over 64 channels.
// Writes fp packed f16 at [p][c*129 + j] (exactly wp's column order).
// ---------------------------------------------------------------------------
__global__ void sample_kernel(const float* __restrict__ f,
                              const float* __restrict__ ct_polys,
                              const float* __restrict__ init_polys,
                              const int* __restrict__ img_idx,
                              _Float16* __restrict__ fpH) {
  int gid = blockIdx.x * blockDim.x + threadIdx.x;
  if (gid >= 2048 * 129) return;
  int p = gid / 129, j = gid - p * 129;
  float px, py;
  if (j == 0) { px = ct_polys[p * 2]; py = ct_polys[p * 2 + 1]; }
  else {
    px = init_polys[(p * 128 + (j - 1)) * 2];
    py = init_polys[(p * 128 + (j - 1)) * 2 + 1];
  }
  int b = img_idx[p];
  float x = px - 0.5f, yv = py - 0.5f;
  float x0f = floorf(x), y0f = floorf(yv);
  float wx = x - x0f, wy = yv - y0f;
  int x0 = (int)x0f, y0 = (int)y0f;
  int x1 = x0 + 1, y1 = y0 + 1;
  float vx0 = (x0 >= 0 && x0 < 128) ? 1.0f : 0.0f;
  float vx1 = (x1 >= 0 && x1 < 128) ? 1.0f : 0.0f;
  float vy0 = (y0 >= 0 && y0 < 128) ? 1.0f : 0.0f;
  float vy1 = (y1 >= 0 && y1 < 128) ? 1.0f : 0.0f;
  int cx0 = x0 < 0 ? 0 : (x0 > 127 ? 127 : x0);
  int cx1 = x1 < 0 ? 0 : (x1 > 127 ? 127 : x1);
  int cy0 = y0 < 0 ? 0 : (y0 > 127 ? 127 : y0);
  int cy1 = y1 < 0 ? 0 : (y1 > 127 ? 127 : y1);
  float w00 = (1.0f - wy) * (1.0f - wx) * vy0 * vx0;
  float w01 = (1.0f - wy) * wx * vy0 * vx1;
  float w10 = wy * (1.0f - wx) * vy1 * vx0;
  float w11 = wy * wx * vy1 * vx1;
  const float* fb = f + (size_t)b * 64 * 16384;
  int o00 = cy0 * 128 + cx0, o01 = cy0 * 128 + cx1;
  int o10 = cy1 * 128 + cx0, o11 = cy1 * 128 + cx1;
#pragma unroll 4
  for (int c = 0; c < 64; ++c) {
    const float* fc = fb + c * 16384;
    float v = fc[o00] * w00 + fc[o01] * w01 + fc[o10] * w10 + fc[o11] * w11;
    fpH[(size_t)p * 8256 + c * 129 + j] = (_Float16)v;
  }
}

// ---------------------------------------------------------------------------
// Generic WMMA GEMM: C[M][N] = A[M][K] * BT[N][K]^T, 128x64 tile per block.
// Double-buffered LDS + async Global->LDS staging: one barrier per K-step,
// step ks+1's DMA overlaps step ks's WMMAs.
// MODE 0: store f16 (h1). MODE 1: out = (acc + bias[n])*4 + init[m][n] (f32).
// ---------------------------------------------------------------------------
template <int MODE>
__global__ __launch_bounds__(256)
void gemm_kernel(const _Float16* __restrict__ A, const _Float16* __restrict__ BT,
                 int M, int N, int K,
                 const float* __restrict__ bias, const float* __restrict__ init,
                 _Float16* __restrict__ outH, float* __restrict__ outF) {
  __shared__ _Float16 sA[2][128 * 40];
  __shared__ _Float16 sB[2][64 * 40];
  const int tid = threadIdx.x;
  const int mBlk = blockIdx.y * 128;
  const int nBlk = blockIdx.x * 64;
  const int wave = tid >> 5;
  const int nt = wave & 3;
  const int mh = wave >> 2;
  const int arow0 = tid >> 2, c8 = tid & 3;

  auto stageAB = [&](int ks, int buf) {
#pragma unroll
    for (int i = 0; i < 2; ++i) {             // A: 512 chunks of 8 halfs
      int row = arow0 + 64 * i;
      stage16(&sA[buf][row * 40 + c8 * 8],
              A + (size_t)(mBlk + row) * K + ks * 32 + c8 * 8);
    }
    stage16(&sB[buf][arow0 * 40 + c8 * 8],    // B: 256 chunks (rows 0..63)
            BT + (size_t)(nBlk + arow0) * K + ks * 32 + c8 * 8);
  };

  v8f acc[4] = {v8f{}, v8f{}, v8f{}, v8f{}};
  const int nSteps = K >> 5;
  stageAB(0, 0);
  for (int ks = 0; ks < nSteps; ++ks) {
    stage_wait_sync();
    const int cur = ks & 1;
    if (ks + 1 < nSteps) stageAB(ks + 1, cur ^ 1);
    v16h bfrag = ld_frag(&sB[cur][nt * 16 * 40], 40);
#pragma unroll
    for (int mt = 0; mt < 4; ++mt) {
      v16h a = ld_frag(&sA[cur][(mh * 64 + mt * 16) * 40], 40);
      acc[mt] = wmma32(a, bfrag, acc[mt]);
    }
  }
  const int lane = tid & 31;
  const int nIdx = nBlk + nt * 16 + (lane & 15);
  const int mOff = (lane >> 4) * 8;
  if (MODE == 0) {
#pragma unroll
    for (int mt = 0; mt < 4; ++mt)
#pragma unroll
      for (int r = 0; r < 8; ++r)
        outH[(size_t)(mBlk + mh * 64 + mt * 16 + mOff + r) * N + nIdx] =
            (_Float16)acc[mt][r];
  } else {
    float bv = bias[nIdx];
#pragma unroll
    for (int mt = 0; mt < 4; ++mt)
#pragma unroll
      for (int r = 0; r < 8; ++r) {
        int m = mBlk + mh * 64 + mt * 16 + mOff + r;
        outF[(size_t)m * N + nIdx] =
            (acc[mt][r] + bv) * 4.0f + init[(size_t)m * N + nIdx];
      }
  }
}

// ---------------------------------------------------------------------------
// Launch
// ---------------------------------------------------------------------------
extern "C" void kernel_launch(void* const* d_in, const int* in_sizes, int n_in,
                              void* d_out, int out_size, void* d_ws, size_t ws_size,
                              hipStream_t stream) {
  const float* feature    = (const float*)d_in[0];
  const float* ct_polys   = (const float*)d_in[1];
  const float* init_polys = (const float*)d_in[2];
  const int*   ct_img_idx = (const int*)d_in[3];
  const float* w1 = (const float*)d_in[4];
  const float* b1 = (const float*)d_in[5];
  const float* w2 = (const float*)d_in[6];
  const float* b2 = (const float*)d_in[7];
  const float* wp = (const float*)d_in[8];
  const float* wf = (const float*)d_in[9];
  const float* bf = (const float*)d_in[10];

  char* ws = (char*)d_ws;
  _Float16* featCL = (_Float16*)(ws + 0);          // 16,777,216 B
  _Float16* w1t    = (_Float16*)(ws + 16777216);   //    294,912 B
  _Float16* w2t    = (_Float16*)(ws + 17072128);   //     32,768 B
  _Float16* wpH    = (_Float16*)(ws + 17104896);   //  8,454,144 B
  _Float16* wfH    = (_Float16*)(ws + 25559040);   //    262,144 B
  _Float16* fpH    = (_Float16*)(ws + 25821184);   // 33,816,576 B
  _Float16* h1H    = (_Float16*)(ws + 59637760);   //  2,097,152 B

  float* coarse = (float*)d_out;                   // 2048*128*2 floats
  float* fOut   = (float*)d_out + 524288;          // 8*64*128*128 floats

  // Stage 0: packs (f16; weights in BT=[out][in] layout, feature chan-last)
  pack_feat_kernel<<<(8388608 + 255) / 256, 256, 0, stream>>>(feature, featCL);
  pack_w1_kernel<<<(147456 + 255) / 256, 256, 0, stream>>>(w1, w1t);
  f32_to_f16_kernel<<<(16384 + 255) / 256, 256, 0, stream>>>(w2, w2t, 16384);
  f32_to_f16_kernel<<<(4227072 + 255) / 256, 256, 0, stream>>>(wp, wpH, 4227072);
  f32_to_f16_kernel<<<(131072 + 255) / 256, 256, 0, stream>>>(wf, wfH, 131072);

  // Stage 1: fused conv1+relu+conv2 -> f (f32, second output region)
  conv_fused_kernel<<<2048, 256, 0, stream>>>(featCL, w1t, w2t, b1, b2, fOut);

  // Stage 2: bilinear sampling -> fp packed f16
  sample_kernel<<<(2048 * 129 + 255) / 256, 256, 0, stream>>>(
      fOut, ct_polys, init_polys, ct_img_idx, fpH);

  // Stage 3: fc1 h1[2048x512] = fp[2048x8256] @ wp^T
  gemm_kernel<0><<<dim3(512 / 64, 2048 / 128), 256, 0, stream>>>(
      fpH, wpH, 2048, 512, 8256, nullptr, nullptr, h1H, nullptr);

  // Stage 4: fc2 + bias + *STRIDE + init_polys -> coarse_polys (f32)
  gemm_kernel<1><<<dim3(256 / 64, 2048 / 128), 256, 0, stream>>>(
      h1H, wfH, 2048, 256, 512, bf, init_polys, nullptr, coarse);
}
